// dense_crf_2362232013211
// MI455X (gfx1250) — matrix operand order repaired and checked by reading, not compiled, not run on hardware
//
#include <hip/hip_runtime.h>
#include <math.h>

// ---------------------------------------------------------------------------
// Dense CRF mean-field (96x96, 21 classes, 10 iters) for MI455X / gfx1250.
//
// msg[c,i] = 3*s_g[i]*Sum_j Kg[i,j]*(s_g[j]*Q[c,j])
//          +10*s_b[i]*Sum_j Kb[i,j]*(s_b[j]*Q[c,j]);  Q = softmax_c(nU + msg)
//
// K entries generated on the fly (v_exp_f32, constants folded to exp2 space)
// and fed into v_wmma_f32_16x16x32_bf16. B tiles hoisted ahead of the exp
// chain; features double-buffered through LDS with register prefetch so all
// VMEM latency hides under the VALU/trans stream (WMMA is TRANS-tracked and
// co-executes). Working set ~2.6 MB -> fully L2 resident on MI455X.
// ---------------------------------------------------------------------------

#define NPIX   9216      // 96*96
#define HW     96
#define NC     21
#define NCP    32        // padded classes
#define WAVES  8         // waves per block in message kernel
#define LOG_P  (-0.22314355f)   // log(0.8)
#define LOG_N  (-4.6051702f)    // log(0.2/20)

// exp2-space constants (log2(e) = 1.4426950408889634)
#define CG   0.02885390f     // 0.02    * log2(e)   (Gaussian spatial)
#define CBS  0.00180337f     // 0.00125 * log2(e)   (bilateral spatial)
#define SC_C 0.08493218f     // 0.1 * sqrt(0.5*log2(e))  (color pre-scale)

typedef __attribute__((ext_vector_type(16))) __bf16 v16bf;
typedef __attribute__((ext_vector_type(8)))  float  v8f;

// ---- init: pre-scaled color features + separable Gaussian normalizer ------
__global__ void k_init(const float* __restrict__ img,
                       float4* __restrict__ feat,
                       float* __restrict__ s_g) {
  int i = blockIdx.x * blockDim.x + threadIdx.x;
  if (i >= NPIX) return;
  feat[i] = make_float4(img[3*i+0]*SC_C, img[3*i+1]*SC_C, img[3*i+2]*SC_C, 0.f);
  int r = i / HW, c = i - (i / HW) * HW;
  float sr = 0.f, sc = 0.f;
  for (int k = 0; k < HW; ++k) {
    float dr = (float)(r - k); sr += __builtin_amdgcn_exp2f(-CG * dr * dr);
    float dc = (float)(c - k); sc += __builtin_amdgcn_exp2f(-CG * dc * dc);
  }
  s_g[i] = rsqrtf(sr * sc + 1e-20f);
}

// ---- init: bilateral normalizer s_b = 1/sqrt(Kb @ 1) (one N^2 pass) -------
__global__ void k_init_sb(const float4* __restrict__ feat,
                          float* __restrict__ s_b) {
  int i = blockIdx.x * blockDim.x + threadIdx.x;
  if (i >= NPIX) return;
  float4 fi = feat[i];
  float ri = (float)(i / HW), ci = (float)(i - (i / HW) * HW);
  float sum = 0.f;
  int rj = 0, cj = 0;
  for (int j = 0; j < NPIX; ++j) {
    float4 fj = feat[j];                       // uniform across wave -> cached
    float dr = ri - (float)rj, dc = ci - (float)cj;
    float d2s = dr * dr + dc * dc;
    float dx = fi.x - fj.x, dy = fi.y - fj.y, dz = fi.z - fj.z;
    float d2c = dx * dx + dy * dy + dz * dz;   // color already exp2-scaled
    sum += __builtin_amdgcn_exp2f(-CBS * d2s - d2c);
    if (++cj == HW) { cj = 0; ++rj; }
  }
  s_b[i] = rsqrtf(sum + 1e-20f);
}

// ---- softmax + prep of bf16 (s*Q)^T operands, layout [class][pixel] -------
__global__ void k_prep(const int* __restrict__ labels,
                       const float* __restrict__ msg,
                       const float* __restrict__ s_g,
                       const float* __restrict__ s_b,
                       __bf16* __restrict__ qsg,
                       __bf16* __restrict__ qsb,
                       int use_msg) {
  int i = blockIdx.x * blockDim.x + threadIdx.x;
  if (i >= NPIX) return;
  int lab = labels[i];
  float logit[NC];
  float m = -1e30f;
#pragma unroll
  for (int c = 0; c < NC; ++c) {
    float v = (c == lab) ? LOG_P : LOG_N;
    if (use_msg) v += msg[c * NPIX + i];
    logit[c] = v;
    m = fmaxf(m, v);
  }
  float sum = 0.f;
#pragma unroll
  for (int c = 0; c < NC; ++c) { logit[c] = __expf(logit[c] - m); sum += logit[c]; }
  float inv = 1.f / sum;
  float sg = s_g[i] * inv, sb = s_b[i] * inv;
#pragma unroll
  for (int c = 0; c < NC; ++c) {
    qsg[c * NPIX + i] = (__bf16)(logit[c] * sg);
    qsb[c * NPIX + i] = (__bf16)(logit[c] * sb);
  }
#pragma unroll
  for (int c = NC; c < NCP; ++c) {            // zero padded class rows
    qsg[c * NPIX + i] = (__bf16)0.f;
    qsb[c * NPIX + i] = (__bf16)0.f;
  }
}

// ---- fused Gaussian+bilateral message pass: on-the-fly K tiles + WMMA -----
__global__ __launch_bounds__(256) void k_message(
    const float4* __restrict__ feat,
    const __bf16* __restrict__ qsg,
    const __bf16* __restrict__ qsb,
    const float* __restrict__ s_g,
    const float* __restrict__ s_b,
    float* __restrict__ msg) {
  __shared__ float4 sfeat[WAVES][2][32];      // double-buffered wave slices
  int lane = threadIdx.x & 31;
  int w    = threadIdx.x >> 5;
  int tile = blockIdx.x * WAVES + w;          // 576 tiles of 16 pixels
  int ibase = tile * 16;
  int half  = lane >> 4;                      // 0: lanes 0-15, 1: lanes 16-31
  int row   = lane & 15;                      // A row (pixel) == B col (class)
  int i = ibase + row;
  float ri = (float)(i / HW), ci = (float)(i - (i / HW) * HW);
  float4 fi = feat[i];

  v8f accg0 = {}, accg1 = {}, accb0 = {}, accb1 = {};

  float4 cur = feat[lane];                    // prefetch j-block 0
  int rjb = 0, cjb = 0;                       // uniform row/col of jb (SALU)

  for (int jb = 0; jb < NPIX; jb += 32) {
    int slot = (jb >> 5) & 1;
    sfeat[w][slot][lane] = cur;               // stage this block (DS in-order)

    // prefetch next block's features (wraps harmlessly on the last step)
    int njb = (jb + 32 == NPIX) ? 0 : jb + 32;
    cur = feat[njb + lane];

    // issue B tiles first so the exp chain below hides their latency.
    // B (32x16 bf16): lanes 0-15 hold K=jb..jb+15, lanes 16-31 K=+16;
    // column = class.  [class][j] layout -> one contiguous 32B per lane.
    int boff = half << 4;
    const v16bf bg0 = *(const v16bf*)(qsg + (row)      * NPIX + jb + boff);
    const v16bf bg1 = *(const v16bf*)(qsg + (16 + row) * NPIX + jb + boff);
    const v16bf bb0 = *(const v16bf*)(qsb + (row)      * NPIX + jb + boff);
    const v16bf bb1 = *(const v16bf*)(qsb + (16 + row) * NPIX + jb + boff);

    // per-lane position deltas without integer division:
    // j = jb + jloc, jloc = tk + half*16, tk = t in [0,15]
    int   lcB  = cjb + (half << 4);           // col of this lane's first j
    float drA  = ri - (float)rjb;             // if no row wrap
    float drB  = drA - 1.0f;                  // if row wrapped
    float dcA  = ci - (float)lcB;             // col delta base (no wrap)
    float dcAw = dcA + (float)HW;             // col delta base (wrapped)

    // A tiles (16x32 bf16), contiguous K to match B: a[t] holds K = t+half*16
    v16bf ag, ab;
#pragma unroll
    for (int t = 0; t < 16; ++t) {
      int   tk   = t;
      bool  wrap = (lcB + tk) >= HW;          // at most one wrap (lcB<96,tk<24)
      float dr   = wrap ? drB : drA;
      float dc   = (wrap ? dcAw : dcA) - (float)tk;
      float d2s  = dr * dr + dc * dc;
      float4 fj  = sfeat[w][slot][tk + (half << 4)];
      float dx = fi.x - fj.x, dy = fi.y - fj.y, dz = fi.z - fj.z;
      float d2c = dx * dx + dy * dy + dz * dz;          // pre-scaled colors
      ag[t] = (__bf16)__builtin_amdgcn_exp2f(-CG * d2s);
      ab[t] = (__bf16)__builtin_amdgcn_exp2f(-CBS * d2s - d2c);
    }

    accg0 = __builtin_amdgcn_wmma_f32_16x16x32_bf16(false, ag, false, bg0,
                                                    (short)0, accg0, false, false);
    accg1 = __builtin_amdgcn_wmma_f32_16x16x32_bf16(false, ag, false, bg1,
                                                    (short)0, accg1, false, false);
    accb0 = __builtin_amdgcn_wmma_f32_16x16x32_bf16(false, ab, false, bb0,
                                                    (short)0, accb0, false, false);
    accb1 = __builtin_amdgcn_wmma_f32_16x16x32_bf16(false, ab, false, bb1,
                                                    (short)0, accb1, false, false);

    // advance uniform jb row/col tracker (SALU, branchless)
    cjb += 32;
    if (cjb >= HW) { cjb -= HW; ++rjb; }
  }

  // D layout: d[v] -> pixel M = v + half*8, class N = row (tile0) / 16+row
#pragma unroll
  for (int v = 0; v < 8; ++v) {
    int p = ibase + v + (half << 3);
    float sg = 3.0f * s_g[p];
    float sb = 10.0f * s_b[p];
    msg[row * NPIX + p]        = sg * accg0[v] + sb * accb0[v];
    msg[(16 + row) * NPIX + p] = sg * accg1[v] + sb * accb1[v];
  }
}

// ---- final argmax (argmax of Q == argmax of logits) -----------------------
__global__ void k_argmax(const int* __restrict__ labels,
                         const float* __restrict__ msg,
                         float* __restrict__ out) {
  int i = blockIdx.x * blockDim.x + threadIdx.x;
  if (i >= NPIX) return;
  int lab = labels[i];
  float best = -1e30f;
  int bi = 0;
#pragma unroll
  for (int c = 0; c < NC; ++c) {
    float v = ((c == lab) ? LOG_P : LOG_N) + msg[c * NPIX + i];
    if (v > best) { best = v; bi = c; }
  }
  out[i] = (float)bi;
}

extern "C" void kernel_launch(void* const* d_in, const int* in_sizes, int n_in,
                              void* d_out, int out_size, void* d_ws, size_t ws_size,
                              hipStream_t stream) {
  const float* img   = (const float*)d_in[0];   // [96*96*3] f32, 0..255
  const int* labels  = (const int*)d_in[1];     // [9216] int
  float* out         = (float*)d_out;           // [9216] class ids as floats

  // carve workspace (~2.6 MB total), 256B-aligned slices
  char* ws = (char*)d_ws;
  size_t off = 0;
  auto carve = [&](size_t bytes) {
    void* p = ws + off;
    off = (off + bytes + 255) & ~(size_t)255;
    return p;
  };
  float4* feat = (float4*)carve(NPIX * sizeof(float4));
  float*  s_g  = (float*) carve(NPIX * sizeof(float));
  float*  s_b  = (float*) carve(NPIX * sizeof(float));
  float*  msg  = (float*) carve((size_t)NCP * NPIX * sizeof(float));
  __bf16* qsg  = (__bf16*)carve((size_t)NCP * NPIX * sizeof(__bf16));
  __bf16* qsb  = (__bf16*)carve((size_t)NCP * NPIX * sizeof(__bf16));

  dim3 blk(256);
  dim3 grdPix(NPIX / 256);          // 36 blocks, exact
  dim3 grdMsg(NPIX / 16 / WAVES);   // 576 tiles / 8 waves = 72 blocks

  k_init   <<<grdPix, blk, 0, stream>>>(img, feat, s_g);
  k_init_sb<<<grdPix, blk, 0, stream>>>(feat, s_b);
  k_prep   <<<grdPix, blk, 0, stream>>>(labels, msg, s_g, s_b, qsg, qsb, 0);
  for (int it = 0; it < 10; ++it) {
    k_message<<<grdMsg, blk, 0, stream>>>(feat, qsg, qsb, s_g, s_b, msg);
    if (it < 9)
      k_prep <<<grdPix, blk, 0, stream>>>(labels, msg, s_g, s_b, qsg, qsb, 1);
  }
  k_argmax <<<grdPix, blk, 0, stream>>>(labels, msg, out);
}
